// DVDNMixer_1692217115203
// MI455X (gfx1250) — matrix operand order repaired and checked by reading, not compiled, not run on hardware
//
#include <hip/hip_runtime.h>
#include <hip/hip_bf16.h>

typedef _Float16 v16h __attribute__((ext_vector_type(16)));
typedef float    v8f  __attribute__((ext_vector_type(8)));

#define N_AGENTS 8
#define N_ATOM   51
#define NPAIRS   (128*256)      // BS*T = 32768
#define PPWG     16             // pairs per workgroup
#define NCHUNK   17             // 17 chunks * 16 bins = 272 >= 257 bins of 512-pt rDFT
#define PSTR     68             // f16 row stride for P tile in LDS (64 used + pad)
#define FSTR     33             // f32 row stride for spectrum tile
#define GSTR     36             // f16 row stride for product tile
// Fragment-order (pre-swizzled) constant matrices in d_ws:
//   Vf[c][ks][nt][lane][16]  : DFT matrix B-fragments for GEMM1
//   Wf[c][nt][lane][16]      : IDFT+projection B-fragments for GEMM2
#define V_ELEMS  (NCHUNK*2*2*32*16)   // 34816 f16
#define W_ELEMS  (NCHUNK*4*32*16)     // 34816 f16
#define TWO_PI_OVER_512 0.0122718463030851562f

// ---- prep: DFT matrix, written directly in WMMA B-fragment order ----
// B-fragment lane map (16-bit B, 32x16): lane half h holds K=16h..16h+15, col=lane%16.
__global__ void dvdn_prep_V(_Float16* __restrict__ Vg) {
  int idx = blockIdx.x * blockDim.x + threadIdx.x;
  if (idx >= V_ELEMS) return;
  int j    = idx & 15;          // element within lane fragment
  int lane = (idx >> 4) & 31;
  int nt   = (idx >> 9) & 1;    // N-tile
  int ks   = (idx >> 10) & 1;   // K-step
  int c    = idx >> 11;         // chunk
  int k    = ks * 32 + 16 * (lane >> 4) + j;  // PMF atom index (K)
  int col  = nt * 16 + (lane & 15);           // 2*bin_in_chunk + reim
  int jd   = c * 16 + (col >> 1);             // DFT bin
  float val = 0.f;
  if (k <= 50 && jd <= 256) {
    int ph = (jd * k) & 511;                  // exact phase mod 512
    float th = (float)ph * TWO_PI_OVER_512;
    val = (col & 1) ? -__sinf(th) : __cosf(th);
  }
  Vg[idx] = (_Float16)val;
}

// ---- prep: fused real-IDFT x exact C51 projection, in B-fragment order ----
// q[m] = sum_j alpha_j (Fr_j cos(2pi j m/512) - Fi_j sin(...)); alpha=1/512 (j=0,256) else 2/512.
// Projection is exact: atom0 = sum_{m<=175} q[m]; atom a in 1..49 = q[175+a]; atom50 = sum_{m>=225}.
__global__ void dvdn_prep_W(_Float16* __restrict__ Wg) {
  int idx = blockIdx.x * blockDim.x + threadIdx.x;
  if (idx >= W_ELEMS) return;
  int j    = idx & 15;
  int lane = (idx >> 4) & 31;
  int nt   = (idx >> 9) & 3;    // N-tile (atom group)
  int c    = idx >> 11;         // chunk
  int row  = 16 * (lane >> 4) + j;   // K index: 2*bin_in_chunk + reim
  int a    = nt * 16 + (lane & 15);  // atom
  int b    = row >> 1, reim = row & 1;
  int jd   = c * 16 + b;
  float val = 0.f;
  if (a <= 50 && jd <= 256) {
    float alpha = (jd == 0 || jd == 256) ? (1.f / 512.f) : (2.f / 512.f);
    if (a >= 1 && a <= 49) {
      int ph = (jd * (175 + a)) & 511;
      float th = (float)ph * TWO_PI_OVER_512;
      val = reim ? -alpha * __sinf(th) : alpha * __cosf(th);
    } else {
      int lo = (a == 0) ? 0 : 225;
      int hi = (a == 0) ? 175 : 400;
      float s = 0.f;
      for (int m = lo; m <= hi; ++m) {
        int ph = (jd * m) & 511;
        float th = (float)ph * TWO_PI_OVER_512;
        s += reim ? __sinf(th) : __cosf(th);
      }
      val = reim ? -alpha * s : alpha * s;
    }
  }
  Wg[idx] = (_Float16)val;
}

// ---- fused main kernel: DFT-GEMM -> 8-way complex product -> IDFT/proj-GEMM ----
__global__ __launch_bounds__(128)
void dvdn_main(const float* __restrict__ P, const _Float16* __restrict__ Vg,
               const _Float16* __restrict__ Wg, float* __restrict__ out) {
  __shared__ _Float16 Pl[128 * PSTR];  // 16 pairs x 8 agents = 128 rows, K padded to 64
  __shared__ float    Fl[128 * FSTR];  // per-agent spectra chunk (f32)
  __shared__ _Float16 Gl[16 * GSTR];   // 8-agent complex products (f16 for GEMM2 A)

  const int tid  = threadIdx.x;
  const int lane = tid & 31;
  const int wv   = tid >> 5;
  const int n16  = lane & 15;
  const int hh   = lane >> 4;
  const long pair_base = (long)blockIdx.x * PPWG;

  const v16h* __restrict__ Vfv = (const v16h*)Vg;  // fragment-order views
  const v16h* __restrict__ Wfv = (const v16h*)Wg;

  // Stage P: 128 rows x 51 f32 -> f16 (zero-pad K to 64)
  {
    const float* src = P + (pair_base * N_AGENTS + tid) * N_ATOM;
    _Float16* dst = &Pl[tid * PSTR];
    #pragma unroll
    for (int k = 0; k < N_ATOM; ++k) dst[k] = (_Float16)src[k];
    #pragma unroll
    for (int k = N_ATOM; k < PSTR; ++k) dst[k] = (_Float16)0.f;
  }
  __syncthreads();

  // Hoist GEMM1 A-fragments: Pl is chunk-invariant. Wave wv owns M-tiles {2wv, 2wv+1}.
  // A frag 16-bit A 16x32, contiguous k order (same as B): lane half h -> K runs [16h..16h+15].
  v16h af[2][2];
  #pragma unroll
  for (int m2 = 0; m2 < 2; ++m2) {
    int mt = wv * 2 + m2;
    #pragma unroll
    for (int ks = 0; ks < 2; ++ks) {
      const _Float16* arow = &Pl[(mt * 16 + n16) * PSTR + ks * 32 + hh * 16];
      v16h a;
      #pragma unroll
      for (int j = 0; j < 8; ++j) { a[j] = arow[j]; a[8 + j] = arow[8 + j]; }
      af[m2][ks] = a;
    }
  }

  v8f acc2 = {};  // persistent GEMM2 accumulator: 16 pairs x atoms [wv*16 .. wv*16+15]

  for (int c = 0; c < NCHUNK; ++c) {
    // Prefetch next chunk's B-fragment blocks (4KB each; 128 threads x 32B slices)
    if (c + 1 < NCHUNK) {
      __builtin_prefetch((const char*)(Vfv + (c + 1) * 128 + tid), 0, 0);
      __builtin_prefetch((const char*)(Wfv + (c + 1) * 128 + tid), 0, 0);
    }

    // GEMM1: F[128 x 32] = P[128 x 64] x V[64 x 32]
    // B-fragments: direct coalesced global loads (L2-hot, pre-swizzled)
    v16h bf[2][2];
    #pragma unroll
    for (int nt = 0; nt < 2; ++nt)
      #pragma unroll
      for (int ks = 0; ks < 2; ++ks)
        bf[nt][ks] = Vfv[((c * 2 + ks) * 2 + nt) * 32 + lane];

    #pragma unroll
    for (int m2 = 0; m2 < 2; ++m2) {
      v8f acc0 = {}, acc1 = {};
      acc0 = __builtin_amdgcn_wmma_f32_16x16x32_f16(false, af[m2][0], false, bf[0][0],
                                                    (short)0, acc0, false, false);
      acc0 = __builtin_amdgcn_wmma_f32_16x16x32_f16(false, af[m2][1], false, bf[0][1],
                                                    (short)0, acc0, false, false);
      acc1 = __builtin_amdgcn_wmma_f32_16x16x32_f16(false, af[m2][0], false, bf[1][0],
                                                    (short)0, acc1, false, false);
      acc1 = __builtin_amdgcn_wmma_f32_16x16x32_f16(false, af[m2][1], false, bf[1][1],
                                                    (short)0, acc1, false, false);
      int rowbase = (wv * 2 + m2) * 16;
      #pragma unroll
      for (int v = 0; v < 8; ++v) {   // C/D layout: row = v + 8*half, col = lane%16
        float* frow = &Fl[(rowbase + v + 8 * hh) * FSTR];
        frow[n16]      = acc0[v];
        frow[16 + n16] = acc1[v];
      }
    }
    __syncthreads();

    // Pointwise: product of 8 agents' spectra per (pair, bin) -> conv spectrum
    #pragma unroll
    for (int rep = 0; rep < 2; ++rep) {
      int idx = rep * 128 + tid;
      int pt = idx >> 4, b = idx & 15;
      const float* f0 = &Fl[(pt * 8) * FSTR + 2 * b];
      float r = f0[0], i = f0[1];
      #pragma unroll
      for (int ag = 1; ag < 8; ++ag) {
        const float* f = &Fl[(pt * 8 + ag) * FSTR + 2 * b];
        float fr = f[0], fi = f[1];
        float nr = r * fr - i * fi;
        float ni = r * fi + i * fr;
        r = nr; i = ni;
      }
      Gl[pt * GSTR + 2 * b]     = (_Float16)r;
      Gl[pt * GSTR + 2 * b + 1] = (_Float16)i;
    }
    __syncthreads();

    // GEMM2 accumulate: out[16 x 64] += G[16 x 32] x W[32 x 64]; N-tile = wave id
    {
      v16h a, b;
      const _Float16* arow = &Gl[n16 * GSTR + hh * 16];
      #pragma unroll
      for (int j = 0; j < 8; ++j) { a[j] = arow[j]; a[8 + j] = arow[8 + j]; }
      b = Wfv[(c * 4 + wv) * 32 + lane];
      acc2 = __builtin_amdgcn_wmma_f32_16x16x32_f16(false, a, false, b,
                                                    (short)0, acc2, false, false);
    }
    __syncthreads();  // protect Fl/Gl reuse next chunk
  }

  // Write results: atoms 0..50 valid (51..63 are padding)
  int atom = wv * 16 + n16;
  if (atom < N_ATOM) {
    #pragma unroll
    for (int v = 0; v < 8; ++v) {
      long pr = pair_base + v + 8 * hh;
      out[pr * N_ATOM + atom] = acc2[v];
    }
  }
}

extern "C" void kernel_launch(void* const* d_in, const int* in_sizes, int n_in,
                              void* d_out, int out_size, void* d_ws, size_t ws_size,
                              hipStream_t stream) {
  const float* P = (const float*)d_in[0];   // agent_qs_distri [128,256,8,51]
  // d_in[1] (states) is unused by the reference computation.
  float* out = (float*)d_out;               // [32768, 51]
  _Float16* Vg = (_Float16*)d_ws;           // 69632 B, 32B-aligned
  _Float16* Wg = Vg + V_ELEMS;              // 69632 B  (total ws: ~136 KB)

  dvdn_prep_V<<<(V_ELEMS + 255) / 256, 256, 0, stream>>>(Vg);
  dvdn_prep_W<<<(W_ELEMS + 255) / 256, 256, 0, stream>>>(Wg);
  dvdn_main<<<NPAIRS / PPWG, 128, 0, stream>>>(P, Vg, Wg, out);
}
